// MemoryLSTM_10050223473347
// MI455X (gfx1250) — matrix operand order repaired and checked by reading, not compiled, not run on hardware
//
#include <hip/hip_runtime.h>
#include <hip/hip_bf16.h>

// ---------------------------------------------------------------------------
// MemoryLSTM for MI455X (gfx1250, wave32).
//   (1) gates_x = x @ W_ih^T + (b_ih+b_hh): bf16 WMMA GEMM, fp32 accumulate,
//       tiles staged via GLOBAL_LOAD_ASYNC_TO_LDS_B128 when available.
//   (2) 4096-step serial recurrence: 16 persistent WGs, W_hh^T sharded
//       256KB-per-WGP in LDS; ONE grid barrier per step (gates double-buffered
//       by step parity; h/c state replicated per-WG in LDS; memory-table
//       scatter written redundantly with identical values by all WGs).
// ---------------------------------------------------------------------------

#define BATCH   32
#define SEQ     128
#define NTOK    (BATCH * SEQ)      // 4096
#define IDIM    512
#define HDIM    512
#define G4H     (4 * HDIM)         // 2048
#define MSZ     256
#define VOCAB   100000
#define CBASE   (HDIM - MSZ)       // 256

typedef __attribute__((ext_vector_type(16))) __bf16 v16bf;
typedef __attribute__((ext_vector_type(8)))  float  v8f;

#if defined(__gfx1250__) && \
    __has_builtin(__builtin_amdgcn_global_load_async_to_lds_b128) && \
    __has_builtin(__builtin_amdgcn_s_wait_asynccnt)
#define USE_ASYNC_LDS 1
typedef int v4i __attribute__((vector_size(16)));
typedef __attribute__((address_space(1))) v4i g1_v4i;   // global int4*
typedef __attribute__((address_space(3))) v4i l3_v4i;   // LDS int4*
#endif

__device__ inline __bf16 f2bf(float f) {
    unsigned u = __builtin_bit_cast(unsigned, f);
    unsigned r = u + 0x7FFFu + ((u >> 16) & 1u);   // round-to-nearest-even
    unsigned short s = (unsigned short)(r >> 16);
    return __builtin_bit_cast(__bf16, s);
}

__device__ inline float sigf(float x) { return 1.0f / (1.0f + __expf(-x)); }
__device__ inline float tanh_fast(float x) {
    x = fminf(fmaxf(x, -15.0f), 15.0f);
    float e = __expf(-2.0f * x);
    return (1.0f - e) / (1.0f + e);
}

// ----------------------------- convert ------------------------------------
__global__ __launch_bounds__(256) void convert_bf16_kernel(
    const float* __restrict__ x, const float* __restrict__ Wih,
    __bf16* __restrict__ xbf, __bf16* __restrict__ wbf,
    long nx, long nw)
{
    long i = (long)blockIdx.x * 256 + threadIdx.x;
    if (i < nx)           xbf[i]      = f2bf(x[i]);
    else if (i < nx + nw) wbf[i - nx] = f2bf(Wih[i - nx]);
}

// ------------------------- WMMA GEMM: gates_x ------------------------------
// out[m,n] = sum_k A[m,k]*W[n,k] + (bih[n]+bhh[n]);  M=4096, N=2048, K=512
// Block: 256 threads = 8 waves (4x2), block tile 128(M) x 64(N), K-step 32.
__global__ __launch_bounds__(256) void gemm_gatesx_kernel(
    const __bf16* __restrict__ A,    // [4096,512]
    const __bf16* __restrict__ W,    // [2048,512]
    const float*  __restrict__ bih, const float* __restrict__ bhh,
    float* __restrict__ out)         // [4096,2048]
{
    // Row pad = 40 halves (80B): every 8-half chunk is 16B-aligned in LDS
    // (required for async B128 deposits), banks spread for fragment reads.
    __shared__ __bf16 As[128][40];
    __shared__ __bf16 Bs[64][40];

    const int tid  = threadIdx.x;
    const int m0   = blockIdx.y * 128;
    const int n0   = blockIdx.x * 64;
    const int wv   = tid >> 5;
    const int lane = tid & 31;
    const int wm   = wv >> 1;        // 0..3
    const int wn   = wv & 1;         // 0..1
    const int lr   = lane & 15;
    const int lh   = lane >> 4;      // 0 or 1 (half-wave)

    v8f acc[2][2] = {};

    for (int k0 = 0; k0 < IDIM; k0 += 32) {
#if defined(USE_ASYNC_LDS)
        // A tile: 128 rows x 32 halves = 512 x 16B chunks; 2 per thread.
        #pragma unroll
        for (int c = tid; c < 512; c += 256) {
            int row = c >> 2, seg = (c & 3) * 8;
            __builtin_amdgcn_global_load_async_to_lds_b128(
                (g1_v4i*)(A + (long)(m0 + row) * IDIM + k0 + seg),
                (l3_v4i*)&As[row][seg], 0, 0);
        }
        // B tile: 64 rows x 32 halves = 256 x 16B chunks; 1 per thread.
        {
            int row = tid >> 2, seg = (tid & 3) * 8;
            __builtin_amdgcn_global_load_async_to_lds_b128(
                (g1_v4i*)(W + (long)(n0 + row) * IDIM + k0 + seg),
                (l3_v4i*)&Bs[row][seg], 0, 0);
        }
        __builtin_amdgcn_s_wait_asynccnt(0);
#else
        {
            int row = tid >> 1, seg = (tid & 1) * 16;
            const __bf16* src = A + (long)(m0 + row) * IDIM + k0 + seg;
            #pragma unroll
            for (int j = 0; j < 16; ++j) As[row][seg + j] = src[j];
        }
        {
            int row = tid >> 2, seg = (tid & 3) * 8;
            const __bf16* src = W + (long)(n0 + row) * IDIM + k0 + seg;
            #pragma unroll
            for (int j = 0; j < 8; ++j) Bs[row][seg + j] = src[j];
        }
#endif
        __syncthreads();

        // Fragment assembly in contiguous k order, same for A and B (wave32, 16-bit A 16x32 / B 32x16)
        v16bf afr[2], bfr[2];
        #pragma unroll
        for (int i = 0; i < 2; ++i) {
            int mr = wm * 32 + i * 16 + lr;
            #pragma unroll
            for (int j = 0; j < 16; ++j)
                afr[i][j] = As[mr][lh * 16 + j];           // K = lh*16 + j
            int nc = wn * 32 + i * 16 + lr;
            #pragma unroll
            for (int j = 0; j < 16; ++j)
                bfr[i][j] = Bs[nc][lh * 16 + j];           // K = lh*16 + j
        }

        #pragma unroll
        for (int i = 0; i < 2; ++i)
            #pragma unroll
            for (int j = 0; j < 2; ++j)
                acc[i][j] = __builtin_amdgcn_wmma_f32_16x16x32_bf16(
                    false, afr[i], false, bfr[j], (short)0, acc[i][j], false, false);
        __syncthreads();
    }

    // Epilogue: C/D layout -> lane<16: M=r ; lane>=16: M=8+r ; N = lane%16
    #pragma unroll
    for (int i = 0; i < 2; ++i) {
        #pragma unroll
        for (int j = 0; j < 2; ++j) {
            int col  = n0 + wn * 32 + j * 16 + lr;
            float bias = bih[col] + bhh[col];
            int rowb = m0 + wm * 32 + i * 16 + lh * 8;
            #pragma unroll
            for (int r = 0; r < 8; ++r)
                out[(long)(rowb + r) * G4H + col] = acc[i][j][r] + bias;
        }
    }
}

// -------------------------- persistent recurrence ---------------------------
#define RWG      16
#define RTHREADS 128
#define WSTRIDE  516   // 512 + 4 pad: float4-aligned, bank-spread

__device__ inline void grid_barrier(int* cnt, int* gen) {
    __threadfence();
    __syncthreads();
    if (threadIdx.x == 0) {
        int g = __hip_atomic_load(gen, __ATOMIC_ACQUIRE, __HIP_MEMORY_SCOPE_AGENT);
        int p = __hip_atomic_fetch_add(cnt, 1, __ATOMIC_ACQ_REL, __HIP_MEMORY_SCOPE_AGENT);
        if (p == RWG - 1) {
            __hip_atomic_store(cnt, 0, __ATOMIC_RELAXED, __HIP_MEMORY_SCOPE_AGENT);
            __hip_atomic_fetch_add(gen, 1, __ATOMIC_RELEASE, __HIP_MEMORY_SCOPE_AGENT);
        } else {
            while (__hip_atomic_load(gen, __ATOMIC_ACQUIRE, __HIP_MEMORY_SCOPE_AGENT) == g)
                __builtin_amdgcn_s_sleep(2);
        }
    }
    __syncthreads();
}

__global__ __launch_bounds__(RTHREADS) void lstm_recur_kernel(
    const float* __restrict__ gates_x,  // [4096,2048] (bias folded)
    const float* __restrict__ Whh,      // [2048,512] row-major
    const int*   __restrict__ toks,     // [4096]
    float* __restrict__ mem,            // ws copy of mem_table [V,256]
    float* __restrict__ gates2,         // [2][2048], parity double-buffered
    int*   bar,                         // two padded {cnt,gen} instances
    float* __restrict__ out_hs,         // [4096,512]
    float* __restrict__ out_last)       // [32,512]
{
    extern __shared__ float smem[];
    float* Wl = smem;                   // [128][WSTRIDE] ~258KB
    float* hl = smem + 128 * WSTRIDE;   // [512]  per-WG replicated h state
    float* cl = hl + HDIM;              // [256]  per-WG replicated c[:base]

    const int nl = threadIdx.x;         // 0..127
    const int wg = blockIdx.x;          // 0..15
    const int ng = wg * RTHREADS + nl;  // gate-output column this thread owns

    // Load this WGP's 128-column slice of W_hh^T into LDS (coalesced on k).
    for (int r = 0; r < 128; ++r) {
        const float* src = Whh + (long)(wg * 128 + r) * HDIM;
        float4 v = *(const float4*)(src + nl * 4);
        *(float4*)(Wl + r * WSTRIDE + nl * 4) = v;
    }
    __syncthreads();

    const float* wp = Wl + nl * WSTRIDE;

    for (int t = 0; t < NTOK; ++t) {
        const int w = t & (SEQ - 1);
        const int p = t & 1;

        // ---- GEMV: gates[ng] = gx[t,ng] + sum_k h[k]*Whh[ng,k]  (h==0 at w==0)
        float acc = gates_x[(long)t * G4H + ng];
        if (t + 1 < NTOK)
            __builtin_prefetch(&gates_x[(long)(t + 1) * G4H + ng], 0, 1);
        if (w != 0) {
            #pragma unroll 4
            for (int k = 0; k < HDIM; k += 4) {
                float4 wv = *(const float4*)(wp + k);
                float4 hv = *(const float4*)(hl + k);
                acc += wv.x * hv.x + wv.y * hv.y + wv.z * hv.z + wv.w * hv.w;
            }
        }
        gates2[(long)p * G4H + ng] = acc;

        grid_barrier(bar + p * 16, bar + p * 16 + 1);   // ONE barrier per step

        // ---- Elementwise: every WG redundantly computes ALL 512 hidden units,
        // keeping h/c in its own LDS so no second barrier is needed.
        {
            const int tok = toks[t];
            const float* gsrc = gates2 + (long)p * G4H;
            #pragma unroll
            for (int q = 0; q < 4; ++q) {
                int j = nl + q * RTHREADS;
                float gi = gsrc[j];
                float gf = gsrc[HDIM + j];
                float gg = gsrc[2 * HDIM + j];
                float go = gsrc[3 * HDIM + j];
                float cprev;
                if (j < CBASE) cprev = (w == 0) ? 0.0f : cl[j];
                else           cprev = mem[(long)tok * MSZ + (j - CBASE)];
                float iv = sigf(gi), fv = sigf(gf);
                float gv = tanh_fast(gg), ov = sigf(go);
                float c2 = fv * cprev + iv * gv;
                float h2 = ov * tanh_fast(c2);
                hl[j] = h2;
                if (j < CBASE) cl[j] = c2;
                else mem[(long)tok * MSZ + (j - CBASE)] = c2;  // identical-value
                                                               // redundant writes
                if (wg == 0) {
                    out_hs[(long)t * HDIM + j] = h2;
                    if (w == SEQ - 1) out_last[(long)(t >> 7) * HDIM + j] = h2;
                }
            }
        }
        __syncthreads();   // hl/cl ready for next step's GEMV
    }
}

// ------------------------------- launch -------------------------------------
extern "C" void kernel_launch(void* const* d_in, const int* in_sizes, int n_in,
                              void* d_out, int out_size, void* d_ws, size_t ws_size,
                              hipStream_t stream) {
    const float* x      = (const float*)d_in[0];
    const int*   toks   = (const int*)  d_in[1];
    const float* Wih    = (const float*)d_in[2];
    const float* Whh    = (const float*)d_in[3];
    const float* bih    = (const float*)d_in[4];
    const float* bhh    = (const float*)d_in[5];
    const float* memtab = (const float*)d_in[6];

    float* out_hs   = (float*)d_out;
    float* out_last = out_hs + (size_t)NTOK * HDIM;

    char* ws = (char*)d_ws;
    size_t off = 0;
    float*  mem  = (float*)(ws + off);  off += (size_t)VOCAB * MSZ * 4;   // 102.4 MB
    float*  gx   = (float*)(ws + off);  off += (size_t)NTOK * G4H * 4;    // 32 MB
    __bf16* xbf  = (__bf16*)(ws + off); off += (size_t)NTOK * IDIM * 2;   // 4 MB
    __bf16* wbf  = (__bf16*)(ws + off); off += (size_t)G4H * IDIM * 2;    // 2 MB
    float*  gts2 = (float*)(ws + off);  off += (size_t)2 * G4H * 4;
    int*    bar  = (int*)(ws + off);    off += 128;

    // Deterministic per-launch init (ws is not re-poisoned between replays).
    (void)hipMemsetAsync(bar, 0, 128, stream);
    // Reference mutates its memory table; inputs are read-only -> private copy.
    (void)hipMemcpyAsync(mem, memtab, (size_t)VOCAB * MSZ * 4,
                         hipMemcpyDeviceToDevice, stream);

    long nx = (long)NTOK * IDIM, nw = (long)G4H * IDIM;
    long tot = nx + nw;
    convert_bf16_kernel<<<(int)((tot + 255) / 256), 256, 0, stream>>>(
        x, Wih, xbf, wbf, nx, nw);

    dim3 g(G4H / 64, NTOK / 128);   // 32 x 32 tiles
    gemm_gatesx_kernel<<<g, 256, 0, stream>>>(xbf, wbf, bih, bhh, gx);

    size_t sh = (size_t)(128 * WSTRIDE + HDIM + MSZ) * sizeof(float);  // ~261 KB
    (void)hipFuncSetAttribute((const void*)lstm_recur_kernel,
                              hipFuncAttributeMaxDynamicSharedMemorySize, (int)sh);
    lstm_recur_kernel<<<RWG, RTHREADS, sh, stream>>>(
        gx, Whh, toks, mem, gts2, bar, out_hs, out_last);
}
